// AFD_22840636080687
// MI455X (gfx1250) — matrix operand order repaired and checked by reading, not compiled, not run on hardware
//
#include <hip/hip_runtime.h>
#include <hip/hip_bf16.h>
#include <math.h>

typedef __attribute__((ext_vector_type(16))) _Float16 v16h;
typedef __attribute__((ext_vector_type(8)))  float    v8f;

#define TEMP_INV 0.5f
#define EPS_BN 1e-5f
#define EPS_LN 1e-5f

struct PtrPack16 { const float* p[16]; };

// ---------------- reduction helpers (wave32) ----------------
__device__ __forceinline__ float waveRedSum(float v) {
#pragma unroll
    for (int o = 16; o > 0; o >>= 1) v += __shfl_xor(v, o, 32);
    return v;
}
__device__ __forceinline__ float waveRedMax(float v) {
#pragma unroll
    for (int o = 16; o > 0; o >>= 1) v = fmaxf(v, __shfl_xor(v, o, 32));
    return v;
}
// blockDim.x == 256 required; returns broadcast value
__device__ __forceinline__ float blockRedSum(float v, float* red) {
    v = waveRedSum(v);
    int lane = threadIdx.x & 31, wid = threadIdx.x >> 5;
    __syncthreads();
    if (lane == 0) red[wid] = v;
    __syncthreads();
    float r = (threadIdx.x < 8) ? red[threadIdx.x] : 0.f;
    if (wid == 0) r = waveRedSum(r);
    if (threadIdx.x == 0) red[0] = r;
    __syncthreads();
    return red[0];
}
__device__ __forceinline__ float blockRedMax(float v, float* red) {
    v = waveRedMax(v);
    int lane = threadIdx.x & 31, wid = threadIdx.x >> 5;
    __syncthreads();
    if (lane == 0) red[wid] = v;
    __syncthreads();
    float r = (threadIdx.x < 8) ? red[threadIdx.x] : -INFINITY;
    if (wid == 0) r = waveRedMax(r);
    if (threadIdx.x == 0) red[0] = r;
    __syncthreads();
    return red[0];
}

// ---------------- student stats: cm, P, pm ----------------
// grid = 64*16 blocks (one per (b,s)), 256 threads. Slab = [16 ch][32x32].
__global__ void k_student(PtrPack16 gs, float* __restrict__ P,
                          float* __restrict__ pm, float* __restrict__ cm) {
    int bs_ = blockIdx.x;
    int b = bs_ >> 4, s = bs_ & 15;
    const float* src = gs.p[s] + (size_t)b * 16 * 1024;
    int t = threadIdx.x;
    float csum[16], cmax[16];
#pragma unroll
    for (int c = 0; c < 16; c++) { csum[c] = 0.f; cmax[c] = -INFINITY; }
    float pmpart = 0.f;
#pragma unroll
    for (int j = 0; j < 4; j++) {
        int p = t + 256 * j;
        float ssq = 0.f;
#pragma unroll
        for (int c = 0; c < 16; c++) {
            float v = src[c * 1024 + p];
            ssq += v * v;
            csum[c] += v;
            cmax[c] = fmaxf(cmax[c], v);
        }
        float pv = ssq * (1.f / 16.f);
        P[(size_t)bs_ * 1024 + p] = pv;
        pmpart += pv;
    }
    __shared__ float red[8];
#pragma unroll
    for (int c = 0; c < 16; c++) {
        float sv = blockRedSum(csum[c], red);
        float mv = blockRedMax(cmax[c], red);
        if (t == 0) cm[bs_ * 16 + c] = 0.7f * (sv * (1.f / 1024.f)) + 0.3f * mv;
    }
    float pt = blockRedSum(pmpart, red);
    if (t == 0) pm[bs_] = pt * (1.f / 1024.f);
}

// ---------------- value_s (adaptive dual-pool + L2 norm), all 4 scales ----------------
__global__ void k_value(const float* __restrict__ P, const float* __restrict__ pm,
                        const float* __restrict__ attn_w, const float* __restrict__ attn_b,
                        float* __restrict__ value_s) {
    int bs_ = blockIdx.x;
    int t = threadIdx.x;
    __shared__ float sp[1024];
    __shared__ float red[8];
#pragma unroll
    for (int j = 0; j < 4; j++) sp[t + 256 * j] = P[(size_t)bs_ * 1024 + t + 256 * j];
    float pmv = pm[bs_];
    __syncthreads();
    const int HT[4] = {32, 16, 8, 4};
    size_t vbase = 0;
    for (int i = 0; i < 4; i++) {
        int Ht = HT[i], N = Ht * Ht, f = 32 / Ht;
        float a = 1.f / (1.f + __expf(-(attn_w[i] * pmv + attn_b[i])));
        float winv = 1.f / (float)(f * f);
        float cvv[4];
        float ssq_part = 0.f;
        for (int j = 0; j < 4; j++) {
            int o = t + 256 * j;
            cvv[j] = 0.f;
            if (o < N) {
                int oy = o / Ht, ox = o % Ht;
                float sum = 0.f, mx = -INFINITY;
                for (int dy = 0; dy < f; dy++)
                    for (int dx = 0; dx < f; dx++) {
                        float v = sp[(oy * f + dy) * 32 + ox * f + dx];
                        sum += v; mx = fmaxf(mx, v);
                    }
                float cv = a * (0.7f * sum * winv + 0.3f * mx);
                cvv[j] = cv;
                ssq_part += cv * cv;
            }
        }
        float ssq = blockRedSum(ssq_part, red);
        float inv = 1.f / fmaxf(sqrtf(ssq), 1e-12f);
        for (int j = 0; j < 4; j++) {
            int o = t + 256 * j;
            if (o < N) value_s[vbase + (size_t)bs_ * N + o] = cvv[j] * inv;
        }
        vbase += (size_t)1024 * N;   // 64*16 slabs of N
    }
}

// ---------------- teacher GAP: one wave per channel ----------------
__global__ void k_gap(const float* __restrict__ ft, float* __restrict__ tmean,
                      int C, int HW) {
    int chunks = C >> 3;
    int b = blockIdx.x / chunks, ch = blockIdx.x % chunks;
    int w = threadIdx.x >> 5, lane = threadIdx.x & 31;
    int c = ch * 8 + w;
    const float* src = ft + (size_t)b * C * HW + (size_t)c * HW;
    float s = 0.f;
    for (int p = lane; p < HW; p += 32) s += src[p];
    s = waveRedSum(s);
    if (lane == 0) tmean[(size_t)b * C + c] = s / (float)HW;
}

// ---------------- teacher h_t: squared channel-mean, L2 normalized ----------------
__global__ void k_ht(const float* __restrict__ ft, float* __restrict__ h_t,
                     int C, int HW) {
    int b = blockIdx.x, t = threadIdx.x;
    __shared__ float buf[1024];
    __shared__ float red[8];
    const float* src = ft + (size_t)b * C * HW;
    float ssq_part = 0.f;
    for (int p = t; p < HW; p += 256) {
        float acc = 0.f;
        for (int c = 0; c < C; c++) { float v = src[(size_t)c * HW + p]; acc += v * v; }
        float raw = acc / (float)C;
        buf[p] = raw;
        ssq_part += raw * raw;
    }
    float ssq = blockRedSum(ssq_part, red);
    float inv = 1.f / fmaxf(sqrtf(ssq), 1e-12f);
    for (int p = t; p < HW; p += 256) h_t[(size_t)b * HW + p] = buf[p] * inv;
}

// ---------------- zero-pad stage-k operands to K=32 ----------------
// Ap[16][64][32] from cm[64][16][16]; Bp[16][128][32] from key_W[16][128][16]
__global__ void k_pad(const float* __restrict__ cm, const float* __restrict__ key_W,
                      float* __restrict__ Ap, float* __restrict__ Bp) {
    int idx = blockIdx.x * 256 + threadIdx.x;
    if (idx < 32768) {
        int k = idx & 31, m = (idx >> 5) & 63, s = idx >> 11;
        Ap[idx] = (k < 16) ? cm[m * 256 + s * 16 + k] : 0.f;
    } else {
        int j = idx - 32768;          // 0 .. 65535
        int k = j & 31, n = (j >> 5) & 127, s = j >> 12;
        Bp[j] = (k < 16) ? key_W[s * 2048 + n * 16 + k] : 0.f;
    }
}

// ---------------- generic WMMA GEMM: C[M,N] = A[M,K] @ Bw^T ----------------
// A[M][K] row-major (lda, multiple of 32); Bw[N][K] row-major (ldb, multiple of 32).
// K % 32 == 0, M,N % 16 == 0. One wave per 16x16 C tile.
// Wide unconditional loads: per K-step, 4x b128 for A, 4x b128 for B.
__global__ void k_gemm(const float* __restrict__ A, int lda,
                       const float* __restrict__ Bw, int ldb,
                       float* __restrict__ C, int ldc, int M, int N, int K) {
    int tn = N >> 4;
    int ntiles = (M >> 4) * tn;
    int tile = blockIdx.x * blockDim.y + threadIdx.y;
    if (tile >= ntiles) return;            // whole-wave uniform exit
    int mt = tile / tn, nt = tile % tn;
    int lane = threadIdx.x;
    int lr = lane & 15, half = lane >> 4;
    int row = mt * 16 + lr;                // A row for this lane
    int col = nt * 16 + lr;                // B/C column for this lane
    // A layout: lane's K values = [16*half, 16*half+16) contiguous in A row `row`
    const float4* ap = (const float4*)(A + (size_t)row * lda + 16 * half);
    // B layout: lane's K values = [16*half, 16*half+16) contiguous in Bw row `col`
    const float4* bp = (const float4*)(Bw + (size_t)col * ldb + 16 * half);
    v8f acc = {};
#pragma unroll 2
    for (int kb = 0; kb < K; kb += 32) {
        float4 a0 = ap[0], a1 = ap[1];     // k = kb+16h .. +7
        float4 a2 = ap[2], a3 = ap[3];     // k = kb+16h+8 .. +7
        float4 b0 = bp[0], b1 = bp[1], b2 = bp[2], b3 = bp[3];
        float af[16] = {a0.x, a0.y, a0.z, a0.w, a1.x, a1.y, a1.z, a1.w,
                        a2.x, a2.y, a2.z, a2.w, a3.x, a3.y, a3.z, a3.w};
        float bf[16] = {b0.x, b0.y, b0.z, b0.w, b1.x, b1.y, b1.z, b1.w,
                        b2.x, b2.y, b2.z, b2.w, b3.x, b3.y, b3.z, b3.w};
        v16h av, bv;
#pragma unroll
        for (int e = 0; e < 16; e++) {
            av[e] = (_Float16)af[e];
            bv[e] = (_Float16)bf[e];
        }
        acc = __builtin_amdgcn_wmma_f32_16x16x32_f16(
            false, av, false, bv, (short)0, acc, false, false);
        ap += 8;                           // 32 floats forward
        bp += 8;
    }
#pragma unroll
    for (int r = 0; r < 8; r++) {
        int rr = mt * 16 + r + 8 * half;   // C/D: VGPR r -> M = r + 8*half
        C[(size_t)rr * ldc + col] = acc[r];
    }
}

// ---------------- BatchNorm(+ReLU) over rows, one wave per column ----------------
__global__ void k_bn(float* __restrict__ X, const float* __restrict__ g,
                     const float* __restrict__ beta, int M, int N, int relu) {
    int w = threadIdx.x >> 5, lane = threadIdx.x & 31;
    int n = blockIdx.x * 8 + w;
    if (n >= N) return;
    float s = 0.f, sq = 0.f;
    for (int r = lane; r < M; r += 32) {
        float v = X[(size_t)r * N + n];
        s += v; sq += v * v;
    }
    s = waveRedSum(s); sq = waveRedSum(sq);
    float mu = s / (float)M;
    float var = sq / (float)M - mu * mu;
    float sc = rsqrtf(var + EPS_BN) * g[n];
    float sh = beta[n];
    for (int r = lane; r < M; r += 32) {
        float v = (X[(size_t)r * N + n] - mu) * sc + sh;
        if (relu) v = fmaxf(v, 0.f);
        X[(size_t)r * N + n] = v;
    }
}

// ---------------- fused cosine-attention + per-teacher loss, one block per batch ----------------
__global__ void k_loss(const float* __restrict__ h2,      // [1024][512] bilinear key
                       const float* __restrict__ query,   // [4][64][128]
                       const float* __restrict__ p_t, const float* __restrict__ p_s,
                       const float* __restrict__ lw,
                       const float* __restrict__ ln_g, const float* __restrict__ ln_b,
                       const float* __restrict__ value_s, const float* __restrict__ h_t,
                       float* __restrict__ partial) {
    int b = blockIdx.x, tid = threadIdx.x;
    int w = tid >> 5, lane = tid & 31;
    __shared__ float knrm[64], qnrm[4], lg[64], entv[4], wts[4], red[8];

    // inverse L2 norms of 64 key vectors and 4 query vectors (len 128)
    for (int V = w; V < 68; V += 8) {
        const float* vp;
        if (V < 64) { int s = V >> 2, tt = V & 3; vp = h2 + (size_t)(b * 16 + s) * 512 + tt * 128; }
        else        { vp = query + (size_t)(V - 64) * 64 * 128 + (size_t)b * 128; }
        float ss = 0.f;
        for (int q = lane; q < 128; q += 32) { float x = vp[q]; ss += x * x; }
        ss = waveRedSum(ss);
        float inv = 1.f / fmaxf(sqrtf(ss), 1e-12f);
        if (lane == 0) { if (V < 64) knrm[V] = inv; else qnrm[V - 64] = inv; }
    }
    __syncthreads();
    // logits = (cos + p_t@p_s^T) / TEMP
    for (int idx = w; idx < 64; idx += 8) {
        int tt = idx >> 4, s = idx & 15;
        const float* kp = h2 + (size_t)(b * 16 + s) * 512 + tt * 128;
        const float* qp = query + (size_t)tt * 64 * 128 + (size_t)b * 128;
        float d = 0.f, pp = 0.f;
        for (int q = lane; q < 128; q += 32) {
            d += kp[q] * qp[q];
            pp += p_t[tt * 128 + q] * p_s[s * 128 + q];
        }
        d = waveRedSum(d); pp = waveRedSum(pp);
        if (lane == 0) lg[tt * 16 + s] = (d * knrm[(s << 2) + tt] * qnrm[tt] + pp) * TEMP_INV;
    }
    __syncthreads();
    // softmax over s (per t) + entropy; lg becomes atts
    if (tid < 4) {
        int tt = tid;
        float m = -INFINITY;
        for (int s = 0; s < 16; s++) m = fmaxf(m, lg[tt * 16 + s]);
        float Z = 0.f;
        for (int s = 0; s < 16; s++) Z += __expf(lg[tt * 16 + s] - m);
        float ent = 0.f;
        for (int s = 0; s < 16; s++) {
            float a = __expf(lg[tt * 16 + s] - m) / Z;
            lg[tt * 16 + s] = a;
            ent += -a * __logf(a + 1e-8f);
        }
        entv[tt] = ent;
    }
    if (tid == 32) {   // softmax(layer_weights)
        float m = fmaxf(fmaxf(lw[0], lw[1]), fmaxf(lw[2], lw[3]));
        float e0 = __expf(lw[0] - m), e1 = __expf(lw[1] - m);
        float e2 = __expf(lw[2] - m), e3 = __expf(lw[3] - m);
        float Z = e0 + e1 + e2 + e3;
        wts[0] = e0 / Z; wts[1] = e1 / Z; wts[2] = e2 / Z; wts[3] = e3 / Z;
    }
    __syncthreads();

    const int NSv[4] = {1024, 256, 64, 16};
    size_t vb = 0, hb = 0;
    float total = 0.f;                       // meaningful at tid 0
    for (int i = 0; i < 4; i++) {
        int N = NSv[i];
        float lng[16], lnb_[16], smA[16], msA[16];
#pragma unroll
        for (int s = 0; s < 16; s++) {
            lng[s] = ln_g[i * 16 + s]; lnb_[s] = ln_b[i * 16 + s];
            smA[s] = 0.f; msA[s] = 0.f;
        }
        const float* vp = value_s + vb + (size_t)b * 16 * N;
        const float* hp = h_t + hb + (size_t)b * N;
        for (int p = tid; p < N; p += 256) {
            float xv[16], mu = 0.f;
#pragma unroll
            for (int s = 0; s < 16; s++) { xv[s] = vp[(size_t)s * N + p]; mu += xv[s]; }
            mu *= (1.f / 16.f);
            float var = 0.f;
#pragma unroll
            for (int s = 0; s < 16; s++) { float dd = xv[s] - mu; var += dd * dd; }
            var *= (1.f / 16.f);
            float rst = rsqrtf(var + EPS_LN);
            float htv = hp[p];
#pragma unroll
            for (int s = 0; s < 16; s++) {
                float hn = (xv[s] - mu) * rst * lng[s] + lnb_[s];
                float d = hn - htv, ad = fabsf(d);
                smA[s] += (ad < 1.f) ? 0.5f * d * d : ad - 0.5f;
                msA[s] += d * d;
            }
        }
        float loss_i = 0.f;
#pragma unroll
        for (int s = 0; s < 16; s++) {
            float smT = blockRedSum(smA[s], red);
            float msT = blockRedSum(msA[s], red);
            if (tid == 0) {
                float smM = smT / (float)N, mseM = msT / (float)N;
                float diff = 0.7f * smM + 0.3f * mseM * mseM;
                loss_i += diff * lg[i * 16 + s];
            }
        }
        if (tid == 0) total += loss_i * wts[i];
        vb += (size_t)64 * 16 * N;
        hb += (size_t)64 * N;
    }
    if (tid == 0) {
        float ent = 0.1f * (entv[0] + entv[1] + entv[2] + entv[3]) * 0.25f;
        partial[b] = (ent + total) * (1.f / 64.f);
    }
}

__global__ void k_final(const float* __restrict__ partial, float* __restrict__ out) {
    int t = threadIdx.x;                 // 32 threads
    float v = partial[t] + partial[t + 32];
    v = waveRedSum(v);
    if (t == 0) out[0] = v;
}

// ---------------- host launch ----------------
static void launch_gemm(const float* A, int lda, const float* Bw, int ldb,
                        float* C, int ldc, int M, int N, int K, hipStream_t stream) {
    int ntiles = (M / 16) * (N / 16);
    dim3 blk(32, 4);
    int nblocks = (ntiles + 3) / 4;
    k_gemm<<<nblocks, blk, 0, stream>>>(A, lda, Bw, ldb, C, ldc, M, N, K);
}

extern "C" void kernel_launch(void* const* d_in, const int* in_sizes, int n_in,
                              void* d_out, int out_size, void* d_ws, size_t ws_size,
                              hipStream_t stream) {
    (void)in_sizes; (void)n_in; (void)out_size; (void)ws_size;
    const float* gt[4];
    for (int i = 0; i < 4; i++) gt[i] = (const float*)d_in[16 + i];
    const float* attn_w   = (const float*)d_in[20];
    const float* attn_b   = (const float*)d_in[21];
    const float* key_W    = (const float*)d_in[22];
    // biases key_b(23), b1(27), b2(31), q_b(38) cancel inside BatchNorm -> skipped
    const float* key_g    = (const float*)d_in[24];
    const float* key_beta = (const float*)d_in[25];
    const float* W1       = (const float*)d_in[26];
    const float* g1       = (const float*)d_in[28];
    const float* beta1    = (const float*)d_in[29];
    const float* W2       = (const float*)d_in[30];
    const float* g2       = (const float*)d_in[32];
    const float* beta2    = (const float*)d_in[33];
    const float* qW[4];
    for (int i = 0; i < 4; i++) qW[i] = (const float*)d_in[34 + i];
    const float* q_g      = (const float*)d_in[39];
    const float* q_beta   = (const float*)d_in[40];
    const float* p_t      = (const float*)d_in[41];
    const float* p_s      = (const float*)d_in[42];
    const float* lw       = (const float*)d_in[43];
    const float* ln_g     = (const float*)d_in[44];
    const float* ln_b     = (const float*)d_in[45];

    float* ws = (float*)d_ws;
    float* P       = ws;                   // 64*16*1024      = 1048576
    float* pm      = P + 1048576;          // 64*16           = 1024
    float* cm      = pm + 1024;            // 64*16*16        = 16384
    float* kbuf    = cm + 16384;           // [64][16][128]   = 131072
    float* h1      = kbuf + 131072;        // 1024*256        = 262144
    float* h2      = h1 + 262144;          // 1024*512        = 524288
    float* tmean   = h2 + 524288;          // 64*(512+1024+2048+2048) = 360448
    float* query   = tmean + 360448;       // 4*64*128        = 32768
    float* ht      = query + 32768;        // 64*(1024+256+64+16) = 87040
    float* vs      = ht + 87040;           // 64*16*1360      = 1392640
    float* partial = vs + 1392640;         // 64
    float* Ap      = partial + 64;         // 16*64*32        = 32768
    float* Bp      = Ap + 32768;           // 16*128*32       = 65536

    PtrPack16 pk;
    for (int i = 0; i < 16; i++) pk.p[i] = (const float*)d_in[i];

    // student descriptors + energy maps
    k_student<<<1024, 256, 0, stream>>>(pk, P, pm, cm);
    // attention-gated adaptive pooling, all scales, normalized
    k_value<<<1024, 256, 0, stream>>>(P, pm, attn_w, attn_b, vs);

    const int TC[4]  = {512, 1024, 2048, 2048};
    const int THW[4] = {1024, 256, 64, 16};
    size_t toff = 0, hoff = 0;
    for (int i = 0; i < 4; i++) {
        k_gap<<<64 * (TC[i] / 8), 256, 0, stream>>>(gt[i], tmean + toff, TC[i], THW[i]);
        k_ht <<<64, 256, 0, stream>>>(gt[i], ht + hoff, TC[i], THW[i]);
        toff += (size_t)64 * TC[i];
        hoff += (size_t)64 * THW[i];
    }

    // pad stage-k operands to K=32, then per-student-layer key projection (WMMA)
    k_pad<<<(32768 + 65536) / 256, 256, 0, stream>>>(cm, key_W, Ap, Bp);
    for (int s = 0; s < 16; s++)
        launch_gemm(Ap + s * 64 * 32, 32, Bp + s * 128 * 32, 32,
                    kbuf + s * 128, 2048, 64, 128, 32, stream);
    k_bn<<<2048 / 8, 256, 0, stream>>>(kbuf, key_g, key_beta, 64, 2048, 1);

    // bilinear-key MLP (WMMA)
    launch_gemm(kbuf, 128, W1, 128, h1, 256, 1024, 256, 128, stream);
    k_bn<<<256 / 8, 256, 0, stream>>>(h1, g1, beta1, 1024, 256, 1);
    launch_gemm(h1, 256, W2, 256, h2, 512, 1024, 512, 256, stream);
    k_bn<<<512 / 8, 256, 0, stream>>>(h2, g2, beta2, 1024, 512, 1);

    // teacher query projections (WMMA) + BN (no relu)
    toff = 0;
    for (int i = 0; i < 4; i++) {
        launch_gemm(tmean + toff, TC[i], qW[i], TC[i],
                    query + (size_t)i * 8192, 128, 64, 128, TC[i], stream);
        k_bn<<<128 / 8, 256, 0, stream>>>(query + (size_t)i * 8192,
                                          q_g + i * 128, q_beta + i * 128, 64, 128, 0);
        toff += (size_t)64 * TC[i];
    }

    // fused attention + loss, then deterministic final reduction
    k_loss<<<64, 256, 0, stream>>>(h2, query, p_t, p_s, lw, ln_g, ln_b, vs, ht, partial);
    k_final<<<1, 32, 0, stream>>>(partial, (float*)d_out);
}
